// TransformerBlock_1408749273335
// MI455X (gfx1250) — matrix operand order repaired and checked by reading, not compiled, not run on hardware
//
#include <hip/hip_runtime.h>
#include <hip/hip_bf16.h>
#include <math.h>

// ---------------------------------------------------------------------------
// Types for CDNA5 WMMA (wave32): 16x16x32 bf16 -> f32
// ---------------------------------------------------------------------------
typedef __attribute__((ext_vector_type(16))) __bf16       bf16x16;
typedef __attribute__((ext_vector_type(8)))  float        f32x8;
typedef __attribute__((ext_vector_type(4)))  unsigned int u32x4;

union Frag  { u32x4 u[2]; bf16x16 v; };   // 32B: one WMMA A or B operand

constexpr int cB = 4, cS = 2048, cD = 512, cH = 8, cDK = 64, cFF = 2048;
constexpr int cBS = cB * cS;  // 8192 token rows

__device__ __forceinline__ f32x8 wmma_bf16(bf16x16 a, bf16x16 b, f32x8 c) {
  // (neg_a, A, neg_b, B, c_mod, C, reuse_a, reuse_b)
  return __builtin_amdgcn_wmma_f32_16x16x32_bf16(false, a, false, b, (short)0, c,
                                                 false, false);
}

// ---------------------------------------------------------------------------
// LayerNorm (f32 in) -> bf16 out.  One block per row, D=512, 2 elems/thread.
// ---------------------------------------------------------------------------
__global__ void __launch_bounds__(256)
ln_bf16_kernel(const float* __restrict__ x, const float* __restrict__ g,
               const float* __restrict__ b, __bf16* __restrict__ out) {
  const int row = blockIdx.x;
  const int t   = threadIdx.x;
  const float* xr = x + (size_t)row * cD;
  float v0 = xr[t], v1 = xr[t + 256];
  __shared__ float red[256];
  red[t] = v0 + v1;
  __syncthreads();
  for (int s = 128; s > 0; s >>= 1) { if (t < s) red[t] += red[t + s]; __syncthreads(); }
  const float mu = red[0] * (1.0f / cD);
  __syncthreads();
  const float d0 = v0 - mu, d1 = v1 - mu;
  red[t] = d0 * d0 + d1 * d1;
  __syncthreads();
  for (int s = 128; s > 0; s >>= 1) { if (t < s) red[t] += red[t + s]; __syncthreads(); }
  const float rs = rsqrtf(red[0] * (1.0f / cD) + 1e-6f);
  __bf16* orow = out + (size_t)row * cD;
  orow[t]       = (__bf16)(d0 * rs * g[t] + b[t]);
  orow[t + 256] = (__bf16)(d1 * rs * g[t + 256] + b[t + 256]);
}

// ---------------------------------------------------------------------------
// Fused f32 -> bf16 cast + transpose: W [K,N] f32 -> Wt [N,K] bf16.
// (weight columns become contiguous-K rows -> WMMA B-fragments are straight
//  16B global loads; weights stay L2-resident on the 192MB L2)
// ---------------------------------------------------------------------------
template <int Kd, int Nd>
__global__ void castT_f32_bf16_kernel(const float* __restrict__ in,
                                      __bf16* __restrict__ out) {
  int i = blockIdx.x * 256 + threadIdx.x;  // over Kd*Nd (exact multiple of 256)
  int n = i % Nd, k = i / Nd;              // coalesced read
  out[(size_t)n * Kd + k] = (__bf16)in[i];
}

// ---------------------------------------------------------------------------
// V [B,S,H*DK] bf16 -> Vt [B,H,DK,S] bf16 (so P*V B-fragments are contiguous)
// ---------------------------------------------------------------------------
__global__ void transpose_v_kernel(const __bf16* __restrict__ v,
                                   __bf16* __restrict__ vt) {
  int i = blockIdx.x * 256 + threadIdx.x;  // over cB*cS*cD (exact multiple)
  int d = i % cDK;
  int h = (i / cDK) % cH;
  int s = (i / cD) % cS;
  int b = i / (cS * cD);
  vt[(((size_t)b * cH + h) * cDK + d) * cS + s] = v[i];
}

// ---------------------------------------------------------------------------
// LDS-free bf16 WMMA GEMM:  C[M,N] = A[M,K] * Wt[N,K]^T  (+ epilogue)
//   Both WMMA operands are per-lane-contiguous-K -> direct 16B global loads.
//   block: 256 thr = 8 waves (4 M x 2 N), block tile 128x128,
//   wave tile 32x64 (2x4 accumulators, 8 WMMA : 12 loads per K-step of 32).
//   EPI 0: bf16 out            EPI 1: +bias, exact GELU, bf16 out
//   EPI 2: +bias +residual, f32 out
// ---------------------------------------------------------------------------
template <int EPI, int N, int K>
__global__ void __launch_bounds__(256)
gemm_bf16_kernel(const __bf16* __restrict__ A, const __bf16* __restrict__ Bt,
                 const float* __restrict__ bias, const float* __restrict__ resid,
                 float* __restrict__ outF, __bf16* __restrict__ outB) {
  const int t    = threadIdx.x;
  const int lane = t & 31, wave = t >> 5;
  const int l16  = lane & 15, lh = lane >> 4;
  const int wm   = wave >> 1, wn = wave & 1;
  const int m0   = blockIdx.x * 128 + wm * 32;
  const int n0   = blockIdx.y * 128 + wn * 64;

  const __bf16* arow = A  + (size_t)(m0 + l16) * K;  // A frag rows (ms adds 16*K)
  const __bf16* bcol = Bt + (size_t)(n0 + l16) * K;  // B frag cols (ns adds 16*K)

  f32x8 acc[2][4] = {};

  for (int k0 = 0; k0 < K; k0 += 32) {
    Frag aF[2], bF[4];
#pragma unroll
    for (int ms = 0; ms < 2; ++ms) {  // A: lane row m=l16, K-window by lane-half
      const __bf16* p = arow + (size_t)ms * 16 * K + k0 + lh * 8;
      aF[ms].u[0] = *(const u32x4*)&p[0];
      aF[ms].u[1] = *(const u32x4*)&p[16];
    }
#pragma unroll
    for (int ns = 0; ns < 4; ++ns) {  // B: lane col n=l16, K-window by lane-half
      const __bf16* p = bcol + (size_t)ns * 16 * K + k0 + lh * 8;
      bF[ns].u[0] = *(const u32x4*)&p[0];
      bF[ns].u[1] = *(const u32x4*)&p[16];
    }
    if (k0 + 32 < K) {  // speculative prefetch of next K-step (global_prefetch_b8)
      __builtin_prefetch((const void*)(arow + k0 + 32), 0, 1);
      __builtin_prefetch((const void*)(bcol + k0 + 32), 0, 1);
    }
#pragma unroll
    for (int ms = 0; ms < 2; ++ms)
#pragma unroll
      for (int ns = 0; ns < 4; ++ns)
        acc[ms][ns] = wmma_bf16(aF[ms].v, bF[ns].v, acc[ms][ns]);
  }

  // Epilogue.  C layout: lane l, VGPR r -> (m = (l>>4)*8 + r, n = l&15)
#pragma unroll
  for (int ms = 0; ms < 2; ++ms)
#pragma unroll
    for (int ns = 0; ns < 4; ++ns)
#pragma unroll
      for (int r = 0; r < 8; ++r) {
        int gr = m0 + ms * 16 + lh * 8 + r;
        int gc = n0 + wn * 0 + ns * 16 + l16;  // n0 already includes wn*64
        float v = acc[ms][ns][r];
        size_t o = (size_t)gr * N + gc;
        if constexpr (EPI == 0) {
          outB[o] = (__bf16)v;
        } else if constexpr (EPI == 1) {
          v += bias[gc];
          v = 0.5f * v * (1.0f + erff(v * 0.70710678118654752f));
          outB[o] = (__bf16)v;
        } else {
          v += bias[gc] + resid[o];
          outF[o] = v;
        }
      }
}

// ---------------------------------------------------------------------------
// Flash attention: 1 wave per (b, h, 16-query tile); stream keys 32 at a time.
// q,k in [B,S,D] bf16 (head slice is contiguous DK=64); vt in [B,H,DK,S] bf16.
// ---------------------------------------------------------------------------
__global__ void __launch_bounds__(32)
attn_kernel(const __bf16* __restrict__ q, const __bf16* __restrict__ k,
            const __bf16* __restrict__ vt, __bf16* __restrict__ ctx) {
  constexpr int LDP = 40;            // padded (80B) P-tile row stride
  __shared__ __bf16 ldsP[16 * LDP];  // 16x32 P tile, C-layout -> A-layout relay
  const int lane = threadIdx.x & 31;
  const int l16  = lane & 15, lh = lane >> 4;
  const int q0   = blockIdx.x * 16;
  const int h    = blockIdx.y;
  const int b    = blockIdx.z;

  // Q A-fragments: 16 x 64 split into two K=32 WMMA steps (contiguous loads)
  Frag aQ[2];
  {
    const __bf16* qp = q + ((size_t)b * cS + q0 + l16) * cD + h * cDK;
#pragma unroll
    for (int s = 0; s < 2; ++s) {
      aQ[s].u[0] = *(const u32x4*)&qp[s * 32 + lh * 8];
      aQ[s].u[1] = *(const u32x4*)&qp[s * 32 + lh * 8 + 16];
    }
  }

  float mrow[8], lrow[8];
#pragma unroll
  for (int r = 0; r < 8; ++r) { mrow[r] = -3.0e38f; lrow[r] = 0.0f; }
  f32x8 o[4] = {};  // 16 x 64 output accumulator

  for (int j0 = 0; j0 < cS; j0 += 32) {
    // ---- scores: two 16x16 tiles (keys j0..15, j0+16..31), contract DK=64
    f32x8 s0 = {}, s1 = {};
#pragma unroll
    for (int st = 0; st < 2; ++st) {
      Frag bK0, bK1;  // B frag: col = key (l16), K-window of DK by lane-half
      const __bf16* kp0 =
          k + ((size_t)b * cS + j0 + l16) * cD + h * cDK + st * 32 + lh * 8;
      const __bf16* kp1 = kp0 + (size_t)16 * cD;
      bK0.u[0] = *(const u32x4*)&kp0[0];
      bK0.u[1] = *(const u32x4*)&kp0[16];
      bK1.u[0] = *(const u32x4*)&kp1[0];
      bK1.u[1] = *(const u32x4*)&kp1[16];
      s0 = wmma_bf16(aQ[st].v, bK0.v, s0);
      s1 = wmma_bf16(aQ[st].v, bK1.v, s1);
    }

    // ---- online softmax (rows live across 16 lanes; butterfly reduce)
#pragma unroll
    for (int r = 0; r < 8; ++r) {
      float a0 = s0[r] * 0.125f, a1 = s1[r] * 0.125f;  // 1/sqrt(DK)
      float pm = fmaxf(a0, a1);
#pragma unroll
      for (int off = 8; off > 0; off >>= 1)
        pm = fmaxf(pm, __shfl_xor(pm, off, 16));
      float mnew = fmaxf(mrow[r], pm);
      float sc   = __expf(mrow[r] - mnew);
      float p0   = __expf(a0 - mnew);
      float p1   = __expf(a1 - mnew);
      float ps   = p0 + p1;
#pragma unroll
      for (int off = 8; off > 0; off >>= 1)
        ps += __shfl_xor(ps, off, 16);
      lrow[r] = lrow[r] * sc + ps;
      mrow[r] = mnew;
#pragma unroll
      for (int ns = 0; ns < 4; ++ns) o[ns][r] *= sc;
      int m = lh * 8 + r;  // C-layout row -> stage P as bf16 in LDS
      ldsP[m * LDP + l16]      = (__bf16)p0;
      ldsP[m * LDP + 16 + l16] = (__bf16)p1;
    }
    __syncthreads();
    Frag aP;  // re-read P in A-fragment layout
    aP.u[0] = *(const u32x4*)&ldsP[l16 * LDP + lh * 8];
    aP.u[1] = *(const u32x4*)&ldsP[l16 * LDP + lh * 8 + 16];
    __syncthreads();

    // ---- o += P (16x32) * V (32x64); Vt gives contiguous B fragments
    const __bf16* vp = vt + ((size_t)b * cH + h) * cDK * cS + j0 + lh * 8;
#pragma unroll
    for (int ns = 0; ns < 4; ++ns) {
      Frag bV;
      const __bf16* c = vp + (size_t)(ns * 16 + l16) * cS;
      bV.u[0] = *(const u32x4*)&c[0];
      bV.u[1] = *(const u32x4*)&c[16];
      o[ns] = wmma_bf16(aP.v, bV.v, o[ns]);
    }
  }

  // ---- normalize & write ctx [B,S,D] bf16
#pragma unroll
  for (int r = 0; r < 8; ++r) {
    float rcp = 1.0f / lrow[r];
    int row = q0 + lh * 8 + r;
#pragma unroll
    for (int ns = 0; ns < 4; ++ns)
      ctx[((size_t)b * cS + row) * cD + h * cDK + ns * 16 + l16] =
          (__bf16)(o[ns][r] * rcp);
  }
}

// ---------------------------------------------------------------------------
// Host: pipeline launch
// ---------------------------------------------------------------------------
extern "C" void kernel_launch(void* const* d_in, const int* in_sizes, int n_in,
                              void* d_out, int out_size, void* d_ws, size_t ws_size,
                              hipStream_t stream) {
  const float* x   = (const float*)d_in[0];
  const float* Wq  = (const float*)d_in[1];
  const float* Wk  = (const float*)d_in[2];
  const float* Wv  = (const float*)d_in[3];
  const float* Wo  = (const float*)d_in[4];
  const float* bo  = (const float*)d_in[5];
  const float* g1  = (const float*)d_in[6];
  const float* b1  = (const float*)d_in[7];
  const float* g2  = (const float*)d_in[8];
  const float* b2  = (const float*)d_in[9];
  const float* W1  = (const float*)d_in[10];
  const float* bf1 = (const float*)d_in[11];
  const float* W2  = (const float*)d_in[12];
  const float* bf2 = (const float*)d_in[13];
  float* out = (float*)d_out;

  char* ws = (char*)d_ws;
  size_t off = 0;
  auto alloc = [&](size_t bytes) -> void* {
    void* p = ws + off;
    off += (bytes + 255) & ~(size_t)255;
    return p;
  };
  __bf16* xn   = (__bf16*)alloc((size_t)cBS * cD * 2);
  __bf16* wqb  = (__bf16*)alloc((size_t)cD * cD * 2);   // transposed [N][K]
  __bf16* wkb  = (__bf16*)alloc((size_t)cD * cD * 2);
  __bf16* wvb  = (__bf16*)alloc((size_t)cD * cD * 2);
  __bf16* wob  = (__bf16*)alloc((size_t)cD * cD * 2);
  __bf16* w1b  = (__bf16*)alloc((size_t)cD * cFF * 2);  // [FF][D]
  __bf16* w2b  = (__bf16*)alloc((size_t)cFF * cD * 2);  // [D][FF]
  __bf16* qb   = (__bf16*)alloc((size_t)cBS * cD * 2);
  __bf16* kb   = (__bf16*)alloc((size_t)cBS * cD * 2);
  __bf16* vb   = (__bf16*)alloc((size_t)cBS * cD * 2);
  __bf16* vtb  = (__bf16*)alloc((size_t)cBS * cD * 2);
  __bf16* ctxb = (__bf16*)alloc((size_t)cBS * cD * 2);
  float*  x1   = (float*)alloc((size_t)cBS * cD * 4);
  __bf16* xn2  = (__bf16*)alloc((size_t)cBS * cD * 2);
  __bf16* hb   = (__bf16*)alloc((size_t)cBS * cFF * 2);

  // 1) pre-norm LN1 -> bf16
  ln_bf16_kernel<<<cBS, 256, 0, stream>>>(x, g1, b1, xn);
  // 2) weight cast+transpose -> [N][K] bf16
  castT_f32_bf16_kernel<cD, cD><<<(cD * cD) / 256, 256, 0, stream>>>(Wq, wqb);
  castT_f32_bf16_kernel<cD, cD><<<(cD * cD) / 256, 256, 0, stream>>>(Wk, wkb);
  castT_f32_bf16_kernel<cD, cD><<<(cD * cD) / 256, 256, 0, stream>>>(Wv, wvb);
  castT_f32_bf16_kernel<cD, cD><<<(cD * cD) / 256, 256, 0, stream>>>(Wo, wob);
  castT_f32_bf16_kernel<cD, cFF><<<(cD * cFF) / 256, 256, 0, stream>>>(W1, w1b);
  castT_f32_bf16_kernel<cFF, cD><<<(cFF * cD) / 256, 256, 0, stream>>>(W2, w2b);
  // 3) Q/K/V projections (LDS-free WMMA GEMM)
  dim3 blk(256);
  dim3 gD(cBS / 128, cD / 128);
  gemm_bf16_kernel<0, cD, cD><<<gD, blk, 0, stream>>>(xn, wqb, nullptr, nullptr, nullptr, qb);
  gemm_bf16_kernel<0, cD, cD><<<gD, blk, 0, stream>>>(xn, wkb, nullptr, nullptr, nullptr, kb);
  gemm_bf16_kernel<0, cD, cD><<<gD, blk, 0, stream>>>(xn, wvb, nullptr, nullptr, nullptr, vb);
  // 4) V -> Vt [B,H,DK,S]
  transpose_v_kernel<<<(cBS * cD) / 256, 256, 0, stream>>>(vb, vtb);
  // 5) flash attention -> ctx bf16
  attn_kernel<<<dim3(cS / 16, cH, cB), 32, 0, stream>>>(qb, kb, vtb, ctxb);
  // 6) output proj + bias + residual -> x1 (f32)
  gemm_bf16_kernel<2, cD, cD><<<gD, blk, 0, stream>>>(ctxb, wob, bo, x, x1, nullptr);
  // 7) LN2 -> bf16
  ln_bf16_kernel<<<cBS, 256, 0, stream>>>(x1, g2, b2, xn2);
  // 8) FFN1 + bias + exact GELU -> bf16
  gemm_bf16_kernel<1, cFF, cD><<<dim3(cBS / 128, cFF / 128), blk, 0, stream>>>(
      xn2, w1b, bf1, nullptr, nullptr, hb);
  // 9) FFN2 + bias + residual -> d_out (f32)
  gemm_bf16_kernel<2, cD, cFF><<<gD, blk, 0, stream>>>(hb, w2b, bf2, x1, out, nullptr);
}
